// StrategyDiscriminator_39127152067171
// MI455X (gfx1250) — matrix operand order repaired and checked by reading, not compiled, not run on hardware
//
#include <hip/hip_runtime.h>
#include <hip/hip_bf16.h>
#include <math.h>

#define B_DIM 4096
#define T_DIM 32
#define A_DIM 4
#define OBS_DIM 115
#define H_DIM 64
#define ACT_DIM 19
#define IN_DIM 99   // 23 dense + 4*19 one-hot

typedef __attribute__((ext_vector_type(16))) _Float16 v16h;
typedef __attribute__((ext_vector_type(8)))  float    v8f;
typedef __attribute__((ext_vector_type(8)))  _Float16 h8;
typedef __attribute__((ext_vector_type(4)))  float    f4;

// Build a v16h A/B fragment from two contiguous 8-half (16B) chunks.
static __device__ __forceinline__ v16h ld_frag2(const _Float16* p0, const _Float16* p1) {
    union { v16h v; h8 h[2]; } u;
    u.h[0] = *(const h8*)p0;
    u.h[1] = *(const h8*)p1;
    return u.v;
}

// B-fragment for v_wmma_f32_16x16x32_f16: lane holds column n = lane%16,
// K chunks = kh*32 + {(lane/16)*8 .. +7, 16 + (lane/16)*8 .. +7} (row-major W[row][64]).
static __device__ __forceinline__ v16h ldB(const _Float16* W, int rowbase, int nloc, int g, int kh) {
    const _Float16* p = W + (size_t)(rowbase + nloc) * 64 + kh * 32 + g * 8;
    return ld_frag2(p, p + 16);
}

static __device__ __forceinline__ v8f bcast8(float x) {
    v8f r;
#pragma unroll
    for (int i = 0; i < 8; ++i) r[i] = x;
    return r;
}

#define WMMA_F16(Af, Bf, Cf) \
    __builtin_amdgcn_wmma_f32_16x16x32_f16(false, (Af), false, (Bf), (short)0, (Cf), false, false)

// Hardware v_rcp_f32 instead of IEEE-division expansion (~1 ulp; irrelevant after f16).
static __device__ __forceinline__ float fast_rcp(float x) { return __builtin_amdgcn_rcpf(x); }
static __device__ __forceinline__ float sigm(float x)      { return fast_rcp(1.f + __expf(-x)); }
static __device__ __forceinline__ float tanh_fast(float x) { return 1.f - 2.f * fast_rcp(1.f + __expf(2.f * x)); }

// ---------------------------------------------------------------- weights f32 -> f16
__global__ void __launch_bounds__(256) wconv_kernel(const float* __restrict__ w0,
                                                    const float* __restrict__ w1,
                                                    const float* __restrict__ w2,
                                                    const float* __restrict__ w3,
                                                    _Float16* __restrict__ out) {
    int i = blockIdx.x * blockDim.x + threadIdx.x;
    const int N = 192 * 64;
    if (i >= 4 * N) return;
    int m = i / N, r = i % N;
    const float* src = (m == 0) ? w0 : (m == 1) ? w1 : (m == 2) ? w2 : w3;
    out[(size_t)m * N + r] = (_Float16)src[r];
}

// ---------------------------------------------------------------- features + proj + LN + ReLU
__global__ void __launch_bounds__(64) featproj_kernel(const float* __restrict__ obs,
                                                      const int* __restrict__ act,
                                                      const float* __restrict__ pW,
                                                      const float* __restrict__ pb,
                                                      const float* __restrict__ lng,
                                                      const float* __restrict__ lnb,
                                                      _Float16* __restrict__ xout) {
    int blk = blockIdx.x;            // = b*T + t
    int t = blk % T_DIM;
    int b = blk / T_DIM;
    int tid = threadIdx.x;           // 0..63 = output channel
    __shared__ float sf[32];
    __shared__ int   sa[4];
    __shared__ float sx[64];

    if (tid == 0) {
        const float* ob = obs + (size_t)blk * A_DIM * OBS_DIM;
        float px[4], py[4];
#pragma unroll
        for (int a = 0; a < 4; ++a) { px[a] = ob[a * OBS_DIM + 0]; py[a] = ob[a * OBS_DIM + 1]; }
        float bx = ob[88], by = ob[89];
        float vx = 0.f, vy = 0.f;
        if (t > 0) { const float* obp = ob - A_DIM * OBS_DIM; vx = bx - obp[88]; vy = by - obp[89]; }
        float cx = 0.f, cy = 0.f;
#pragma unroll
        for (int a = 0; a < 4; ++a) { cx += px[a]; cy += py[a]; }
        cx *= 0.25f; cy *= 0.25f;
        float sp = 0.f;
#pragma unroll
        for (int a = 0; a < 4; ++a) {
            float dx = px[a] - cx, dy = py[a] - cy;
            sp += dx * dx + dy * dy;
        }
        sp = sqrtf(sp * 0.25f + 1e-6f);
        sf[0] = bx; sf[1] = by; sf[2] = vx; sf[3] = vy;
#pragma unroll
        for (int a = 0; a < 4; ++a) { sf[4 + 2 * a] = px[a]; sf[5 + 2 * a] = py[a]; }
        sf[12] = cx; sf[13] = cy; sf[14] = sp;
#pragma unroll
        for (int a = 0; a < 4; ++a) {
            float best = 1e30f; int bsel = 0;
#pragma unroll
            for (int o = 0; o < 4; ++o) {
                if (o == a) continue;
                float dx = px[o] - px[a], dy = py[o] - py[a];
                float d2 = dx * dx + dy * dy;
                if (d2 < best) { best = d2; bsel = o; }
            }
            sf[15 + 2 * a] = px[bsel] - px[a];
            sf[16 + 2 * a] = py[bsel] - py[a];
        }
    }
    if (tid < 4) sa[tid] = act[(size_t)blk * 4 + tid];
    __syncthreads();

    const float* w = pW + (size_t)tid * IN_DIM;
    float acc = pb[tid];
#pragma unroll
    for (int k = 0; k < 23; ++k) acc += sf[k] * w[k];
#pragma unroll
    for (int a = 0; a < 4; ++a) acc += w[23 + a * ACT_DIM + sa[a]];   // sparse one-hot
    sx[tid] = acc;
    __syncthreads();

    float s = 0.f, s2 = 0.f;
#pragma unroll 8
    for (int k = 0; k < 64; ++k) { float v = sx[k]; s += v; s2 += v * v; }
    float mu = s * (1.f / 64.f);
    float var = s2 * (1.f / 64.f) - mu * mu;
    float xn = (acc - mu) * rsqrtf(var + 1e-5f) * lng[tid] + lnb[tid];
    xn = fmaxf(xn, 0.f);
    xout[((size_t)t * B_DIM + b) * H_DIM + tid] = (_Float16)xn;
}

// ---------------------------------------------------------------- GRU layer (WMMA recurrence)
// Block = 128 threads (4 waves) per 16-row batch tile. Wave j owns gate-column tile j.
// x_{t+1} fragments are prefetched during step t's compute (x is recurrence-independent).
__global__ void __launch_bounds__(128) gru_kernel(const _Float16* __restrict__ xin,   // [T][B][64] f16
                                                  const _Float16* __restrict__ Wih,   // [192][64] f16
                                                  const _Float16* __restrict__ Whh,   // [192][64] f16
                                                  const float* __restrict__ bih,
                                                  const float* __restrict__ bhh,
                                                  _Float16* __restrict__ ysout,       // [T][B][64] or null
                                                  _Float16* __restrict__ hlast) {     // [B][64] or null
    int bt   = blockIdx.x;
    int tid  = threadIdx.x;
    int wave = tid >> 5;
    int lane = tid & 31;
    int nloc = lane & 15;
    int g    = lane >> 4;
    int j    = wave;                 // column tile 0..3

    __shared__ __align__(16) _Float16 sh[16 * 64];   // h tile, row-major f16

    // Preload B fragments (constant over T): gates r(j), z(j+4), n(j+8); K=64 => 2 halves each.
    v16h BiR0 = ldB(Wih,   0 + j * 16, nloc, g, 0), BiR1 = ldB(Wih,   0 + j * 16, nloc, g, 1);
    v16h BiZ0 = ldB(Wih,  64 + j * 16, nloc, g, 0), BiZ1 = ldB(Wih,  64 + j * 16, nloc, g, 1);
    v16h BiN0 = ldB(Wih, 128 + j * 16, nloc, g, 0), BiN1 = ldB(Wih, 128 + j * 16, nloc, g, 1);
    v16h BhR0 = ldB(Whh,   0 + j * 16, nloc, g, 0), BhR1 = ldB(Whh,   0 + j * 16, nloc, g, 1);
    v16h BhZ0 = ldB(Whh,  64 + j * 16, nloc, g, 0), BhZ1 = ldB(Whh,  64 + j * 16, nloc, g, 1);
    v16h BhN0 = ldB(Whh, 128 + j * 16, nloc, g, 0), BhN1 = ldB(Whh, 128 + j * 16, nloc, g, 1);

    // Per-column biases (C/D layout: column = lane%16, same for all 8 row-regs).
    int nc = j * 16 + nloc;
    float br = bih[nc]       + bhh[nc];
    float bz = bih[64 + nc]  + bhh[64 + nc];
    float bi = bih[128 + nc];
    float bh = bhh[128 + nc];

    v8f hC = {};                     // persistent f32 h tile (C/D layout)
#pragma unroll
    for (int i = 0; i < 8; ++i) sh[tid * 8 + i] = (_Float16)0.f;

    // Prefetch x fragments for t=0.
    const _Float16* xbase = xin + ((size_t)bt * 16 + nloc) * H_DIM;
    const _Float16* xrow  = xbase;                    // + t*B*64 added per step
    v16h xA0 = ld_frag2(xrow      + g * 8, xrow + 16 + g * 8);
    v16h xA1 = ld_frag2(xrow + 32 + g * 8, xrow + 48 + g * 8);
    __syncthreads();

    for (int t = 0; t < T_DIM; ++t) {
        // h A fragment from LDS (lane = row M, K chunks {g*8..},{16+g*8..} per 32-half).
        const _Float16* hrow = sh + nloc * 64;
        v16h hA0 = ld_frag2(hrow      + g * 8, hrow + 16 + g * 8);
        v16h hA1 = ld_frag2(hrow + 32 + g * 8, hrow + 48 + g * 8);
        __syncthreads();             // everyone done reading old h

        v8f ar = bcast8(br), az = bcast8(bz), ai = bcast8(bi), ah = bcast8(bh);
        ar = WMMA_F16(xA0, BiR0, ar); ar = WMMA_F16(xA1, BiR1, ar);
        az = WMMA_F16(xA0, BiZ0, az); az = WMMA_F16(xA1, BiZ1, az);
        ai = WMMA_F16(xA0, BiN0, ai); ai = WMMA_F16(xA1, BiN1, ai);

        // Prefetch next step's x while the h-side WMMAs + gate math run.
        int tn = (t + 1 < T_DIM) ? (t + 1) : (T_DIM - 1);
        const _Float16* xrn = xbase + (size_t)tn * B_DIM * H_DIM;
        v16h nxA0 = ld_frag2(xrn      + g * 8, xrn + 16 + g * 8);
        v16h nxA1 = ld_frag2(xrn + 32 + g * 8, xrn + 48 + g * 8);

        ar = WMMA_F16(hA0, BhR0, ar); ar = WMMA_F16(hA1, BhR1, ar);
        az = WMMA_F16(hA0, BhZ0, az); az = WMMA_F16(hA1, BhZ1, az);
        ah = WMMA_F16(hA0, BhN0, ah); ah = WMMA_F16(hA1, BhN1, ah);

#pragma unroll
        for (int v = 0; v < 8; ++v) {
            float r  = sigm(ar[v]);
            float z  = sigm(az[v]);
            float nn = tanh_fast(ai[v] + r * ah[v]);
            float hn = nn + z * (hC[v] - nn);        // (1-z)*n + z*h
            hC[v] = hn;
            sh[(g * 8 + v) * 64 + j * 16 + nloc] = (_Float16)hn;   // row m=g*8+v, col=j*16+nloc
        }
        __syncthreads();             // new h visible

        if (ysout) {                 // stream this step's h out (layer 0)
            int e = tid * 8, m = e >> 6, c = e & 63;
            *(f4*)(ysout + ((size_t)t * B_DIM + bt * 16 + m) * H_DIM + c) = *(const f4*)(sh + e);
        }

        xA0 = nxA0; xA1 = nxA1;
    }

    if (hlast) {
        int e = tid * 8, m = e >> 6, c = e & 63;
        *(f4*)(hlast + (size_t)(bt * 16 + m) * H_DIM + c) = *(const f4*)(sh + e);
    }
}

// ---------------------------------------------------------------- classifier (tiny)
__global__ void __launch_bounds__(64) cls_kernel(const _Float16* __restrict__ hl,
                                                 const float* __restrict__ W1, const float* __restrict__ b1,
                                                 const float* __restrict__ W2, const float* __restrict__ b2,
                                                 float* __restrict__ out) {
    int b = blockIdx.x, j = threadIdx.x;
    __shared__ float hid[64];
    const _Float16* h = hl + (size_t)b * 64;
    const float* w = W1 + (size_t)j * 64;
    float acc = b1[j];
#pragma unroll 8
    for (int k = 0; k < 64; ++k) acc += (float)h[k] * w[k];
    hid[j] = fmaxf(acc, 0.f);
    __syncthreads();
    if (j < 8) {
        const float* w2 = W2 + (size_t)j * 64;
        float o = b2[j];
#pragma unroll 8
        for (int k = 0; k < 64; ++k) o += hid[k] * w2[k];
        out[(size_t)b * 8 + j] = o;
    }
}

// ---------------------------------------------------------------- launch
extern "C" void kernel_launch(void* const* d_in, const int* in_sizes, int n_in,
                              void* d_out, int out_size, void* d_ws, size_t ws_size,
                              hipStream_t stream) {
    const float* obs  = (const float*)d_in[0];
    const int*   act  = (const int*)d_in[1];
    const float* pW   = (const float*)d_in[2];
    const float* pb   = (const float*)d_in[3];
    const float* lng  = (const float*)d_in[4];
    const float* lnb  = (const float*)d_in[5];
    const float* Wih0 = (const float*)d_in[6];
    const float* Whh0 = (const float*)d_in[7];
    const float* bih0 = (const float*)d_in[8];
    const float* bhh0 = (const float*)d_in[9];
    const float* Wih1 = (const float*)d_in[10];
    const float* Whh1 = (const float*)d_in[11];
    const float* bih1 = (const float*)d_in[12];
    const float* bhh1 = (const float*)d_in[13];
    const float* cW1  = (const float*)d_in[14];
    const float* cb1  = (const float*)d_in[15];
    const float* cW2  = (const float*)d_in[16];
    const float* cb2  = (const float*)d_in[17];
    float* out = (float*)d_out;

    char* ws = (char*)d_ws;
    size_t off = 0;
    _Float16* x0  = (_Float16*)(ws + off); off += (size_t)T_DIM * B_DIM * H_DIM * 2;   // 16 MB
    _Float16* ys0 = (_Float16*)(ws + off); off += (size_t)T_DIM * B_DIM * H_DIM * 2;   // 16 MB
    _Float16* hl  = (_Float16*)(ws + off); off += (size_t)B_DIM * H_DIM * 2;           // 0.5 MB
    _Float16* wh  = (_Float16*)(ws + off); off += (size_t)4 * 192 * 64 * 2;            // 96 KB
    _Float16* Wih0h = wh;
    _Float16* Whh0h = wh + 12288;
    _Float16* Wih1h = wh + 2 * 12288;
    _Float16* Whh1h = wh + 3 * 12288;

    wconv_kernel<<<(4 * 192 * 64 + 255) / 256, 256, 0, stream>>>(Wih0, Whh0, Wih1, Whh1, wh);
    featproj_kernel<<<B_DIM * T_DIM, 64, 0, stream>>>(obs, act, pW, pb, lng, lnb, x0);
    gru_kernel<<<B_DIM / 16, 128, 0, stream>>>(x0,  Wih0h, Whh0h, bih0, bhh0, ys0, nullptr);
    gru_kernel<<<B_DIM / 16, 128, 0, stream>>>(ys0, Wih1h, Whh1h, bih1, bhh1, nullptr, hl);
    cls_kernel<<<B_DIM, 64, 0, stream>>>(hl, cW1, cb1, cW2, cb2, out);
}
